// SinkhornWarpInterpolator_1692217115468
// MI455X (gfx1250) — compile-verified, matrix operand order checked by reading, not run on hardware
//
#include <hip/hip_runtime.h>

typedef __attribute__((ext_vector_type(16))) _Float16 v16h;
typedef __attribute__((ext_vector_type(8)))  float    v8f;

#define B_     16
#define CCH    4
#define H_     256
#define W_     256
#define PATCH  4
#define WIN    8
#define M_     64      // tokens per window (8x8)
#define D_     64      // feature dim (C*P*P)
#define TAU    0.05f
#define ITERS  20
#define LDH    72      // padded f16 row stride (bank-conflict-free: 36 dw % 64)
#define LDF    65      // padded f32 row stride for the 64x64 log-plan

__global__ __launch_bounds__(256) void sinkhorn_warp_fused(
    const float* __restrict__ z0, const float* __restrict__ z1,
    float* __restrict__ out)
{
  __shared__ _Float16 sA[M_ * LDH];   // window-0 tokens [token][feat], f16
  __shared__ _Float16 sB[M_ * LDH];   // window-1 tokens
  __shared__ float    sLP[M_ * LDF];  // logits / log-plan
  __shared__ float    sDX[M_], sDY[M_];

  const int tid = threadIdx.x;
  const int wid = blockIdx.x;            // window id 0..1023
  const int bb  = wid >> 6;              // batch
  const int wh  = (wid >> 3) & 7;        // window row
  const int ww  = wid & 7;               // window col

  // ---------- phase 1: patchify + L2-normalize -> f16 LDS tiles ----------
  // thread t: token i = t/4, sub = t%4 == channel c; 4x4 pixel patch of c
  {
    const int i   = tid >> 2;
    const int sub = tid & 3;
    const int ty  = wh * WIN + (i >> 3);
    const int tx  = ww * WIN + (i & 7);
    const int py0 = ty * PATCH;
    const int px0 = tx * PATCH;
    for (int srcsel = 0; srcsel < 2; ++srcsel) {
      const float* z = srcsel ? z1 : z0;
      _Float16*    s = srcsel ? sB : sA;
      const float* base = z + (((size_t)(bb * CCH + sub)) * H_ + py0) * W_ + px0;
      float v[16];
      float ssq = 0.f;
      #pragma unroll
      for (int r = 0; r < 4; ++r) {
        float4 q = *reinterpret_cast<const float4*>(base + (size_t)r * W_);
        v[r*4+0] = q.x; v[r*4+1] = q.y; v[r*4+2] = q.z; v[r*4+3] = q.w;
        ssq += q.x*q.x + q.y*q.y + q.z*q.z + q.w*q.w;
      }
      // reduce sum of squares over the 4 lanes owning this token
      ssq += __shfl_xor(ssq, 1, 32);
      ssq += __shfl_xor(ssq, 2, 32);
      const float inv = 1.0f / fmaxf(sqrtf(ssq), 1e-6f);
      _Float16* dst = s + i * LDH + sub * 16;
      #pragma unroll
      for (int k = 0; k < 16; ++k) dst[k] = (_Float16)(v[k] * inv);
    }
  }
  __syncthreads();

  // ---------- phase 2: logits = (w0 . w1^T)/tau via v_wmma_f32_16x16x32_f16
  // 4x4 grid of 16x16 tiles; K=64 as two 16x16x32 steps; 8 waves x 2 tiles.
  {
    const int wave = tid >> 5;
    const int lane = tid & 31;
    const int hlf  = lane >> 4;          // lane half selects K+8 per ISA layout
    const int lm   = lane & 15;          // M (A) / N (B) within tile
    const float invTau = 1.0f / TAU;
    #pragma unroll
    for (int tt = 0; tt < 2; ++tt) {
      const int t  = wave * 2 + tt;
      const int ti = t >> 2, tj = t & 3;
      v8f acc = {};
      #pragma unroll
      for (int ks = 0; ks < 2; ++ks) {
        v16h a, bf;
        const _Float16* arow = sA + (ti * 16 + lm) * LDH;
        const _Float16* brow = sB + (tj * 16 + lm) * LDH;
        #pragma unroll
        for (int e = 0; e < 16; ++e) {
          const int vq = e >> 1;
          const int k  = ks * 32 + ((vq >= 4) ? 16 : 0) + hlf * 8
                       + (vq & 3) * 2 + (e & 1);
          a[e]  = arow[k];
          bf[e] = brow[k];
        }
        acc = __builtin_amdgcn_wmma_f32_16x16x32_f16(
            false, a, false, bf, (short)0, acc, false, false);
      }
      #pragma unroll
      for (int r = 0; r < 8; ++r)
        sLP[(ti * 16 + hlf * 8 + r) * LDF + tj * 16 + lm] = acc[r] * invTau;
    }
  }
  __syncthreads();

  // ---------- phase 3: Sinkhorn (log domain), 20 x (row-LSE, col-LSE) ------
  // group g = tid/4 owns row/col g; 4 lanes x 16 elements, shfl_xor reduce.
  const int g   = tid >> 2;
  const int sub = tid & 3;
  const int c0  = sub * 16;
  for (int it = 0; it < ITERS; ++it) {
    { // row pass (reduce over j)
      float* row = sLP + g * LDF;
      float m = -__builtin_inff();
      #pragma unroll
      for (int k = 0; k < 16; ++k) m = fmaxf(m, row[c0 + k]);
      m = fmaxf(m, __shfl_xor(m, 1, 32));
      m = fmaxf(m, __shfl_xor(m, 2, 32));
      float s = 0.f;
      #pragma unroll
      for (int k = 0; k < 16; ++k) s += __expf(row[c0 + k] - m);
      s += __shfl_xor(s, 1, 32);
      s += __shfl_xor(s, 2, 32);
      const float lse = m + __logf(s);
      #pragma unroll
      for (int k = 0; k < 16; ++k) row[c0 + k] -= lse;
    }
    __syncthreads();
    { // col pass (reduce over i)
      float m = -__builtin_inff();
      #pragma unroll
      for (int k = 0; k < 16; ++k) m = fmaxf(m, sLP[(c0 + k) * LDF + g]);
      m = fmaxf(m, __shfl_xor(m, 1, 32));
      m = fmaxf(m, __shfl_xor(m, 2, 32));
      float s = 0.f;
      #pragma unroll
      for (int k = 0; k < 16; ++k) s += __expf(sLP[(c0 + k) * LDF + g] - m);
      s += __shfl_xor(s, 1, 32);
      s += __shfl_xor(s, 2, 32);
      const float lse = m + __logf(s);
      #pragma unroll
      for (int k = 0; k < 16; ++k) sLP[(c0 + k) * LDF + g] -= lse;
    }
    __syncthreads();
  }

  // ---------- phase 4: row softmax -> expected displacement ----------------
  {
    float* row = sLP + g * LDF;
    float m = -__builtin_inff();
    #pragma unroll
    for (int k = 0; k < 16; ++k) m = fmaxf(m, row[c0 + k]);
    m = fmaxf(m, __shfl_xor(m, 1, 32));
    m = fmaxf(m, __shfl_xor(m, 2, 32));
    float s = 0.f, sx = 0.f, sy = 0.f;
    #pragma unroll
    for (int k = 0; k < 16; ++k) {
      const int j = c0 + k;
      const float e = __expf(row[j] - m);
      s  += e;
      sx += e * (float)(j & 7);
      sy += e * (float)(j >> 3);
    }
    s  += __shfl_xor(s, 1, 32);  s  += __shfl_xor(s, 2, 32);
    sx += __shfl_xor(sx, 1, 32); sx += __shfl_xor(sx, 2, 32);
    sy += __shfl_xor(sy, 1, 32); sy += __shfl_xor(sy, 2, 32);
    if (sub == 0) {
      sDX[g] = sx / s - (float)(g & 7);
      sDY[g] = sy / s - (float)(g >> 3);
    }
  }
  __syncthreads();

  // ---------- phase 5: bilinear border warp of this window's 32x32x4 tile --
  {
    for (int p = tid; p < 1024; p += 256) {
      const int ly = p >> 5, lx = p & 31;
      const int y  = wh * 32 + ly;
      const int x  = ww * 32 + lx;
      const int i  = (ly >> 2) * 8 + (lx >> 2);     // owning token in window
      float gx = (float)x + sDX[i] * (float)PATCH;
      float gy = (float)y + sDY[i] * (float)PATCH;
      gx = fminf(fmaxf(gx, 0.f), (float)(W_ - 1));
      gy = fminf(fmaxf(gy, 0.f), (float)(H_ - 1));
      const float x0f = floorf(gx), y0f = floorf(gy);
      const float wx = gx - x0f, wy = gy - y0f;
      const int x0 = (int)x0f, y0 = (int)y0f;
      const int x1 = min(x0 + 1, W_ - 1);
      const int y1 = min(y0 + 1, H_ - 1);
      const float w00 = (1.f - wx) * (1.f - wy);
      const float w01 = wx * (1.f - wy);
      const float w10 = (1.f - wx) * wy;
      const float w11 = wx * wy;
      #pragma unroll
      for (int c = 0; c < CCH; ++c) {
        const float* zc = z1 + ((size_t)(bb * CCH + c)) * H_ * W_;
        const float v = zc[y0 * W_ + x0] * w00 + zc[y0 * W_ + x1] * w01
                      + zc[y1 * W_ + x0] * w10 + zc[y1 * W_ + x1] * w11;
        out[(((size_t)(bb * CCH + c)) * H_ + y) * W_ + x] = v;
      }
    }
  }
}

extern "C" void kernel_launch(void* const* d_in, const int* in_sizes, int n_in,
                              void* d_out, int out_size, void* d_ws, size_t ws_size,
                              hipStream_t stream) {
  (void)in_sizes; (void)n_in; (void)d_ws; (void)ws_size; (void)out_size;
  const float* z0 = (const float*)d_in[0];
  const float* z1 = (const float*)d_in[1];
  float* out = (float*)d_out;
  // 1024 windows (16 batches x 8 x 8), one workgroup (8 wave32) per window
  sinkhorn_warp_fused<<<dim3(1024), dim3(256), 0, stream>>>(z0, z1, out);
}
